// GlobalAttention_19533511262702
// MI455X (gfx1250) — matrix operand order repaired and checked by reading, not compiled, not run on hardware
//
#include <hip/hip_runtime.h>
#include <hip/hip_bf16.h>

// ---- CDNA5 WMMA types ----
typedef __attribute__((ext_vector_type(16))) __bf16        v16bf;
typedef __attribute__((ext_vector_type(8)))  float         v8f;
typedef __attribute__((ext_vector_type(4)))  unsigned int  u32x4;

union BF16x16 { v16bf v; u32x4 q[2]; };

static __device__ __forceinline__ unsigned short f2bf_bits(float f) {
    __bf16 h = (__bf16)f;
    return __builtin_bit_cast(unsigned short, h);
}

// LDS byte-offset of a generic pointer to __shared__ memory (AS3 offset)
static __device__ __forceinline__ unsigned lds_offset_of(const void* p) {
    return (unsigned)(unsigned long long)(__attribute__((address_space(3))) const unsigned char*)p;
}

// Async DMA: 16 bytes global -> LDS per lane, tracked by ASYNCcnt (CDNA5).
// vdst = per-lane LDS byte offset, vaddr = per-lane 64-bit global address.
static __device__ __forceinline__ void async_g2lds_b128(unsigned lds_off, const void* gaddr) {
    asm volatile("global_load_async_to_lds_b128 %0, %1, off"
                 :: "v"(lds_off), "v"((unsigned long long)(size_t)gaddr)
                 : "memory");
}

static __device__ __forceinline__ void wait_asynccnt0() {
#if __has_builtin(__builtin_amdgcn_s_wait_asynccnt)
    __builtin_amdgcn_s_wait_asynccnt(0);
#else
    asm volatile("s_wait_asynccnt 0x0" ::: "memory");
#endif
}

#define NB   32      // B
#define NN   512     // N
#define NL   4096    // L
#define NG   128     // G
#define NDIM 64      // DIM
#define AL   3.0f

// ---------------------------------------------------------------------------
// K1: n{1,2}[row,col] = tanh(3*(emb[idx[row],:] @ lin_w[:,col] + lin_b[col]))
// stored bf16 row-major [512][64]. One thread per output element (2*512*64).
// ---------------------------------------------------------------------------
__global__ __launch_bounds__(256) void k1_nodes(
    const int* __restrict__ idx,
    const float* __restrict__ emb1, const float* __restrict__ emb2,
    const float* __restrict__ w1, const float* __restrict__ b1,
    const float* __restrict__ w2, const float* __restrict__ b2,
    __bf16* __restrict__ n1b, __bf16* __restrict__ n2b)
{
    int tid   = blockIdx.x * blockDim.x + threadIdx.x;   // 65536 threads
    int which = tid >> 15;
    int rem   = tid & 32767;
    int row   = rem >> 6, col = rem & 63;
    const float* emb  = which ? emb2 : emb1;
    const float* w    = which ? w2   : w1;
    const float* bias = which ? b2   : b1;
    int r = idx[row];
    float s = bias[col];
    #pragma unroll 8
    for (int k = 0; k < NDIM; ++k) s = fmaf(emb[r * NDIM + k], w[k * NDIM + col], s);
    float t = tanhf(AL * s);
    (which ? n2b : n1b)[row * NDIM + col] = (__bf16)t;
}

// ---------------------------------------------------------------------------
// K2: adjT[s,l] = tanh(3*((n1 n2^T)[l,s] - (n2 n1^T)[l,s]))
// Two bf16 WMMA accumulators per K-chunk; one wave per 16x16 tile (32x32 tiles).
// ---------------------------------------------------------------------------
__global__ __launch_bounds__(256) void k2_adj(
    const __bf16* __restrict__ n1b, const __bf16* __restrict__ n2b,
    __bf16* __restrict__ adjTb)
{
    int wid  = blockIdx.x * 8 + (threadIdx.x >> 5);      // 1024 waves
    int lane = threadIdx.x & 31;
    int st = wid >> 5, lt = wid & 31;
    int m = lane & 15, hi = lane >> 4;
    const unsigned short* n1u = (const unsigned short*)n1b;
    const unsigned short* n2u = (const unsigned short*)n2b;
    v8f accP = {}; v8f accM = {};
    #pragma unroll
    for (int k0 = 0; k0 < NDIM; k0 += 32) {
        BF16x16 aP, aM, bP, bM;
        // A layout (16-bit 16x32): lane<16 K={0..7,16..23}, lane>=16 K={8..15,24..31}
        int aoff = (st * 16 + m) * NDIM + k0 + hi * 8;
        aP.q[0] = *(const u32x4*)(n2u + aoff); aP.q[1] = *(const u32x4*)(n2u + aoff + 16);
        aM.q[0] = *(const u32x4*)(n1u + aoff); aM.q[1] = *(const u32x4*)(n1u + aoff + 16);
        // B layout (32x16): lane<16 K={0..7,16..23}, lane>=16 K={8..15,24..31}, N=lane%16
        int boff = (lt * 16 + m) * NDIM + k0 + hi * 8;
        bP.q[0] = *(const u32x4*)(n1u + boff); bP.q[1] = *(const u32x4*)(n1u + boff + 16);
        bM.q[0] = *(const u32x4*)(n2u + boff); bM.q[1] = *(const u32x4*)(n2u + boff + 16);
        accP = __builtin_amdgcn_wmma_f32_16x16x32_bf16(false, aP.v, false, bP.v, (short)0, accP, false, false);
        accM = __builtin_amdgcn_wmma_f32_16x16x32_bf16(false, aM.v, false, bM.v, (short)0, accM, false, false);
    }
    #pragma unroll
    for (int r = 0; r < 8; ++r) {
        int s = st * 16 + r + hi * 8;
        int l = lt * 16 + m;
        adjTb[s * NN + l] = (__bf16)tanhf(AL * (accP[r] - accM[r]));
    }
}

// ---------------------------------------------------------------------------
// K3: q[b,n] = tanh(x[b, n>>3, :] . mlp_w[:, n&7] + mlp_b[n&7])
// Only rows 0..63 of each x[b] are used (raw-reshape + [:,0] in reference).
// ---------------------------------------------------------------------------
__global__ __launch_bounds__(256) void k3_q(
    const float* __restrict__ x, const float* __restrict__ mlp_w,
    const float* __restrict__ mlp_b, float* __restrict__ q)
{
    int wid  = blockIdx.x * 8 + (threadIdx.x >> 5);      // 2048 waves
    int lane = threadIdx.x & 31;
    int b = wid >> 6, row = wid & 63;
    const float* xr = x + ((size_t)b * NN + row) * NL;
    float acc[8] = {0,0,0,0,0,0,0,0};
    for (int l = lane; l < NL; l += 32) {
        float xv = xr[l];
        const float* wr = mlp_w + (size_t)l * 8;
        #pragma unroll
        for (int c = 0; c < 8; ++c) acc[c] = fmaf(xv, wr[c], acc[c]);
    }
    #pragma unroll
    for (int c = 0; c < 8; ++c) {
        float s = acc[c];
        #pragma unroll
        for (int off = 16; off >= 1; off >>= 1) s += __shfl_xor(s, off, 32);
        if (lane == 0) q[b * NN + row * 8 + c] = tanhf(s + mlp_b[c]);
    }
}

// ---------------------------------------------------------------------------
// K3b: selT[b][g][l] = relu(q[b,l] * W_sel[0,g,0])  stored bf16 (K-major)
// ---------------------------------------------------------------------------
__global__ __launch_bounds__(256) void k3b_sel(
    const float* __restrict__ q, const float* __restrict__ Wsel,
    __bf16* __restrict__ selT)
{
    int t = blockIdx.x * blockDim.x + threadIdx.x;       // 32*128*512
    int b = t >> 16;
    int g = (t >> 9) & 127;
    int l = t & 511;
    float v = q[b * NN + l] * Wsel[g];                   // h==0 slice
    selT[t] = (__bf16)(v > 0.0f ? v : 0.0f);             // t == ((b*128+g)*512+l)
}

// ---------------------------------------------------------------------------
// K4: scores[b,s,g] = sum_l adjT[s,l] * selT[b,g,l]   (WMMA bf16)
// ---------------------------------------------------------------------------
__global__ __launch_bounds__(256) void k4_scores(
    const __bf16* __restrict__ adjTb, const __bf16* __restrict__ selT,
    float* __restrict__ scores)
{
    int wid  = blockIdx.x * 8 + (threadIdx.x >> 5);      // 8192 waves
    int lane = threadIdx.x & 31;
    int b   = wid >> 8;
    int rem = wid & 255;
    int st = rem >> 3, gt = rem & 7;
    int m = lane & 15, hi = lane >> 4;
    const unsigned short* au = (const unsigned short*)adjTb;
    const unsigned short* su = (const unsigned short*)selT;
    v8f acc = {};
    int srow = st * 16 + m;
    int grow = gt * 16 + m;
    for (int l0 = 0; l0 < NN; l0 += 32) {
        BF16x16 a, bm;
        int aoff = srow * NN + l0 + hi * 8;
        a.q[0]  = *(const u32x4*)(au + aoff);
        a.q[1]  = *(const u32x4*)(au + aoff + 16);
        int boff = (b * NG + grow) * NN + l0 + hi * 8;
        bm.q[0] = *(const u32x4*)(su + boff);
        bm.q[1] = *(const u32x4*)(su + boff + 16);
        acc = __builtin_amdgcn_wmma_f32_16x16x32_bf16(false, a.v, false, bm.v, (short)0, acc, false, false);
    }
    #pragma unroll
    for (int r = 0; r < 8; ++r) {
        int s = st * 16 + r + hi * 8;
        int g = gt * 16 + m;
        scores[((size_t)b * NN + s) * NG + g] = acc[r];
    }
}

// ---------------------------------------------------------------------------
// K5: series = softmax(scores, axis=-1) -> transposed bf16 seriesT[b][g][s]
// ---------------------------------------------------------------------------
__global__ __launch_bounds__(256) void k5_softmax(
    const float* __restrict__ scores, __bf16* __restrict__ seriesT)
{
    int wid  = blockIdx.x * 8 + (threadIdx.x >> 5);      // 16384 waves
    int lane = threadIdx.x & 31;
    int b = wid >> 9, s = wid & 511;
    const float* row = scores + ((size_t)b * NN + s) * NG;
    float v[4];
    float mx = -3.0e38f;
    #pragma unroll
    for (int i = 0; i < 4; ++i) { v[i] = row[lane + 32 * i]; mx = fmaxf(mx, v[i]); }
    #pragma unroll
    for (int off = 16; off >= 1; off >>= 1) mx = fmaxf(mx, __shfl_xor(mx, off, 32));
    float sum = 0.0f;
    #pragma unroll
    for (int i = 0; i < 4; ++i) { v[i] = __expf(v[i] - mx); sum += v[i]; }
    #pragma unroll
    for (int off = 16; off >= 1; off >>= 1) sum += __shfl_xor(sum, off, 32);
    float inv = 1.0f / sum;
    #pragma unroll
    for (int i = 0; i < 4; ++i) {
        int g = lane + 32 * i;
        seriesT[((size_t)b * NG + g) * NN + s] = (__bf16)(v[i] * inv);
    }
}

// ---------------------------------------------------------------------------
// K6: V[b,s,g] = sum_l x[b,l,s] * series[b,l,g]      (the 17.2 GFLOP kernel)
// Workgroup = 128 threads (4 waves), output block = 64(s) x 128(g).
// Per 32-wide l-chunk:
//   - x chunk (f32) -> convert bf16 -> LDS transposed [s][l], padded stride
//   - seriesT chunk (bf16) -> LDS via GLOBAL_LOAD_ASYNC_TO_LDS_B128 (ASYNCcnt)
//   - 8 WMMAs per wave, A and B both from conflict-free ds_load_b128.
// LDS row stride 80B: 16B-aligned and (80/4)=20 mod 64 gives all-distinct
// starting banks for 16 consecutive rows -> no bank conflicts.
// ---------------------------------------------------------------------------
#define A_STRIDE 80                       // bytes per s-row   (32 l as bf16 + pad)
#define B_STRIDE 80                       // bytes per g-row   (32 l as bf16 + pad)
#define A_BYTES  (64 * A_STRIDE)          // 5120
#define B_BYTES  (128 * B_STRIDE)         // 10240

__global__ __launch_bounds__(128) void k6_v(
    const float* __restrict__ x, const __bf16* __restrict__ seriesT,
    float* __restrict__ V)
{
    __shared__ unsigned char smem[A_BYTES + B_BYTES];    // 15 KB
    int b = blockIdx.x >> 6;
    int sBlock = (blockIdx.x & 63) * 64;
    int t = threadIdx.x;
    int w = t >> 5, lane = t & 31;
    int m = lane & 15, hi = lane >> 4;
    const unsigned short* su = (const unsigned short*)seriesT;
    const float* xb = x + (size_t)b * NN * NL;
    unsigned ldsBase  = lds_offset_of(smem);
    unsigned ldsBBase = ldsBase + A_BYTES;

    v8f zero = {};
    v8f acc[8];
    #pragma unroll
    for (int c = 0; c < 8; ++c) acc[c] = zero;

    int sStage = t & 63;
    int jb = t >> 6;                                     // 0 or 1
    for (int l0 = 0; l0 < NN; l0 += 32) {
        // ---- async-stage B: seriesT[b][0..127][l0..l0+31] -> LDS, 8KB, once per WG
        #pragma unroll
        for (int i = 0; i < 4; ++i) {
            int u   = t + 128 * i;                       // 0..511 : (g, 16B-seg)
            int g   = u >> 2;
            int seg = u & 3;
            const void* ga = (const void*)(su + ((size_t)b * NG + g) * NN + l0 + seg * 8);
            async_g2lds_b128(ldsBBase + g * B_STRIDE + seg * 16, ga);
        }
        // ---- stage A: x f32 -> bf16, transposed into LDS [s][l]
        unsigned int* ldsA = (unsigned int*)smem;
        #pragma unroll
        for (int jj = 0; jj < 8; ++jj) {
            int j = jj * 2 + jb;
            int l = l0 + 2 * j;
            const float* p0 = xb + (size_t)l * NL + sBlock + sStage;
            float f0 = p0[0];
            float f1 = p0[NL];
            if (l0 + 32 < NN) __builtin_prefetch(p0 + 32 * NL, 0, 1);  // next chunk
            ldsA[sStage * (A_STRIDE / 4) + j] =
                (unsigned)f2bf_bits(f0) | ((unsigned)f2bf_bits(f1) << 16);
        }
        wait_asynccnt0();
        __syncthreads();
        // ---- compute: wave w owns s-tile w; 8 g-tiles
        BF16x16 a;
        const unsigned char* lbA = smem;
        const unsigned char* lbB = smem + A_BYTES;
        int aoff = (w * 16 + m) * A_STRIDE + hi * 16;
        a.q[0] = *(const u32x4*)(lbA + aoff);
        a.q[1] = *(const u32x4*)(lbA + aoff + 32);
        #pragma unroll
        for (int c = 0; c < 8; ++c) {
            BF16x16 bm;
            int boff = (c * 16 + m) * B_STRIDE + hi * 16;
            bm.q[0] = *(const u32x4*)(lbB + boff);
            bm.q[1] = *(const u32x4*)(lbB + boff + 32);
            acc[c] = __builtin_amdgcn_wmma_f32_16x16x32_bf16(false, a.v, false, bm.v, (short)0, acc[c], false, false);
        }
        __syncthreads();
    }
    #pragma unroll
    for (int c = 0; c < 8; ++c) {
        #pragma unroll
        for (int r = 0; r < 8; ++r) {
            int s = sBlock + w * 16 + r + hi * 8;
            int g = c * 16 + m;
            V[((size_t)b * NL + s) * NG + g] = acc[c][r];
        }
    }
}

// ---------------------------------------------------------------------------
extern "C" void kernel_launch(void* const* d_in, const int* in_sizes, int n_in,
                              void* d_out, int out_size, void* d_ws, size_t ws_size,
                              hipStream_t stream)
{
    (void)in_sizes; (void)n_in; (void)out_size; (void)ws_size;
    const int*   idx    = (const int*)  d_in[0];
    const float* x      = (const float*)d_in[5];
    const float* emb1   = (const float*)d_in[6];
    const float* emb2   = (const float*)d_in[7];
    const float* lin1_w = (const float*)d_in[8];
    const float* lin1_b = (const float*)d_in[9];
    const float* lin2_w = (const float*)d_in[10];
    const float* lin2_b = (const float*)d_in[11];
    const float* mlp_w  = (const float*)d_in[12];
    const float* mlp_b  = (const float*)d_in[13];
    const float* W_sel  = (const float*)d_in[14];
    float* out = (float*)d_out;

    // workspace layout (bf16 intermediates), ~8.7 MB total
    char* ws = (char*)d_ws;
    __bf16* n1b     = (__bf16*)(ws);                       //  64 KB
    __bf16* n2b     = (__bf16*)(ws + 65536);               //  64 KB
    __bf16* adjTb   = (__bf16*)(ws + 131072);              // 512 KB
    float*  q       = (float*) (ws + 655360);              //  64 KB
    __bf16* selT    = (__bf16*)(ws + 720896);              //   4 MB
    __bf16* seriesT = (__bf16*)(ws + 4915200);             //   4 MB
    float*  scores  = out;  // 8 MB scratch inside d_out; fully overwritten by k6

    k1_nodes  <<<256,  256, 0, stream>>>(idx, emb1, emb2, lin1_w, lin1_b, lin2_w, lin2_b, n1b, n2b);
    k2_adj    <<<128,  256, 0, stream>>>(n1b, n2b, adjTb);
    k3_q      <<<256,  256, 0, stream>>>(x, mlp_w, mlp_b, q);
    k3b_sel   <<<8192, 256, 0, stream>>>(q, W_sel, selT);
    k4_scores <<<1024, 256, 0, stream>>>(adjTb, selT, scores);
    k5_softmax<<<2048, 256, 0, stream>>>(scores, seriesT);
    k6_v      <<<2048, 128, 0, stream>>>(x, seriesT, out);
}
